// TankBindTriangleProteinToCompound_90142773608721
// MI455X (gfx1250) — matrix operand order repaired and checked by reading, not compiled, not run on hardware
//
#include <hip/hip_runtime.h>
#include <hip/hip_bf16.h>
#include <stdint.h>

// ---------------------------------------------------------------------------
// TankBind triangle update for MI455X (gfx1250), wave32 + WMMA bf16 path.
// Pipeline: LN+gated-proj (f32 -> bf16 intermediates, WMMA), per-channel
// triangle GEMMs (WMMA, f32 accum, double-buffered async-to-LDS staging),
// fused LN(C)+C->E GEMM epilogue (WMMA).
// ---------------------------------------------------------------------------

typedef __attribute__((ext_vector_type(16))) __bf16 v16bf;
typedef __attribute__((ext_vector_type(8)))  float  v8f;

#define E_DIM 128
#define C_DIM 32
#define NP_DIM 768
#define NC_DIM 128
#define NPTS (NP_DIM * NC_DIM)   // 98304

// --------------------------- CDNA5 async-to-LDS ----------------------------
#if defined(__has_builtin)
#  if __has_builtin(__builtin_amdgcn_global_load_async_to_lds_b128)
#    define HAVE_ASYNC_LDS 1
#  endif
#endif
#ifndef HAVE_ASYNC_LDS
#  define HAVE_ASYNC_LDS 0
#endif

// b128 payload type exactly as the builtin expects it (per hipcc diagnostic):
// "__attribute__((__vector_size__(4 * sizeof(int)))) int"
typedef int v4i_ __attribute__((vector_size(4 * sizeof(int))));
typedef __attribute__((address_space(1))) v4i_ gas_v4i;   // global
typedef __attribute__((address_space(3))) v4i_ las_v4i;   // LDS

__device__ __forceinline__ void async_cp16(const void* gsrc, void* ldst) {
#if HAVE_ASYNC_LDS
  // generic->AS1: global generic address == AS1 address.
  // generic->AS3: low 32 bits of a generic LDS address are the LDS offset
  // (ISA 10.2 aperture rules).
  __builtin_amdgcn_global_load_async_to_lds_b128(
      (gas_v4i*)(uintptr_t)gsrc,
      (las_v4i*)(uint32_t)(uintptr_t)ldst,
      0, 0);
#else
  *(uint32_t*)ldst       = ((const uint32_t*)gsrc)[0];
  ((uint32_t*)ldst)[1]   = ((const uint32_t*)gsrc)[1];
  ((uint32_t*)ldst)[2]   = ((const uint32_t*)gsrc)[2];
  ((uint32_t*)ldst)[3]   = ((const uint32_t*)gsrc)[3];
#endif
}

// Wait until at most N async-to-LDS groups remain in flight (per-wave,
// in-order completion => waiting <=N retires everything older).
template <int N>
__device__ __forceinline__ void wait_async_le() {
#if HAVE_ASYNC_LDS
#  if __has_builtin(__builtin_amdgcn_s_wait_asynccnt)
  __builtin_amdgcn_s_wait_asynccnt(N);
#  else
  asm volatile("s_wait_asynccnt %0" :: "i"(N) : "memory");
#  endif
#endif
}

// ------------------------------- math helpers ------------------------------
__device__ __forceinline__ float fast_rcp(float x) {
#if defined(__has_builtin) && __has_builtin(__builtin_amdgcn_rcpf)
  return __builtin_amdgcn_rcpf(x);
#else
  return 1.0f / x;
#endif
}
__device__ __forceinline__ float sigmoidf_(float x) {
  return fast_rcp(1.0f + __expf(-x));   // v_exp_f32 + v_add + v_rcp_f32
}

// ------------------------------ WMMA fragments -----------------------------
// 16-bit A fragment (16x32, MxK): lane m(0..15) = row m, lanes 16..31 = rows
// 0..15 with K offset +8.  VGPR v<4 -> K = 2v,2v+1 ; v>=4 -> K = 16+2(v-4).
__device__ __forceinline__ int a_kbase(int v, int lane) {
  int khalf = (lane >> 4) & 1;
  int base = (v < 4) ? (2 * v) : (16 + 2 * (v - 4));
  return base + 8 * khalf;
}
// 16-bit B fragment (32x16, KxN): lane n(0..15) = column n; same K layout as A
// (lanes 16..31 K offset +8; v<4 -> K = 2v,2v+1 ; v>=4 -> K = 16+2(v-4)).
__device__ __forceinline__ int b_kbase(int v, int lane) {
  int khalf = (lane >> 4) & 1;
  int base = (v < 4) ? (2 * v) : (16 + 2 * (v - 4));
  return base + 8 * khalf;
}

__device__ __forceinline__ v16bf frag_a_ld(const __bf16* base, int ldk) {
  int lane = threadIdx.x & 31;
  const __bf16* p = base + (size_t)(lane & 15) * ldk;
  v16bf f;
#pragma unroll
  for (int v = 0; v < 8; ++v) {
    int k = a_kbase(v, lane);
    f[2 * v]     = p[k];
    f[2 * v + 1] = p[k + 1];
  }
  return f;
}
__device__ __forceinline__ v16bf frag_b_ld(const __bf16* base, int ldk) {
  int lane = threadIdx.x & 31;
  const __bf16* p = base + (size_t)(lane & 15) * ldk;
  v16bf f;
#pragma unroll
  for (int v = 0; v < 8; ++v) {
    int k = b_kbase(v, lane);
    f[2 * v]     = p[k];
    f[2 * v + 1] = p[k + 1];
  }
  return f;
}

__device__ __forceinline__ v8f wmma_bf16(v16bf a, v16bf b, v8f acc) {
  return __builtin_amdgcn_wmma_f32_16x16x32_bf16(
      /*neg_a=*/false, a, /*neg_b=*/false, b,
      /*c_mod=*/(short)0, acc, /*reuse_a=*/false, /*reuse_b=*/false);
}

// ---------------------------------------------------------------------------
// Kernel 1: LayerNorm(E) + gated linear projection E->C.
//   out[c][x][y] = sig(ln(row).glw[c]) * (ln(row).lw[c] + lb[c]) [* mask]
// row r -> (p, q) = (r / W, r % W); (x, y) = swapxy ? (q, p) : (p, q).
// 128 threads (4 waves), 64 rows per block.
// ---------------------------------------------------------------------------
__global__ void k_ln_gate_proj(const float* __restrict__ in,
                               const float* __restrict__ mask,   // per-row or nullptr
                               const float* __restrict__ lnw,    // (E,)
                               const float* __restrict__ glw,    // (C,E) row-major = [n][k]
                               const float* __restrict__ lw,     // (C,E)
                               const float* __restrict__ lb,     // (C,)
                               __bf16* __restrict__ out,
                               int W, int chanStride, int rowStride, int swapxy,
                               int streamNT) {
  __shared__ __bf16 sA[64][E_DIM];
  __shared__ __bf16 sWg[C_DIM][E_DIM];
  __shared__ __bf16 sWl[C_DIM][E_DIM];
  const int tid = threadIdx.x;
  const int rowBase = blockIdx.x * 64;

  for (int idx = tid; idx < C_DIM * E_DIM; idx += 128) {
    sWg[idx >> 7][idx & 127] = (__bf16)glw[idx];
    sWl[idx >> 7][idx & 127] = (__bf16)lw[idx];
  }
  if (streamNT) {
    // read-once streams (protein_pair): keep them out of L2's working set
    for (int idx = tid; idx < 64 * E_DIM; idx += 128)
      sA[idx >> 7][idx & 127] =
          (__bf16)__builtin_nontemporal_load(&in[(size_t)rowBase * E_DIM + idx]);
  } else {
    for (int idx = tid; idx < 64 * E_DIM; idx += 128)
      sA[idx >> 7][idx & 127] = (__bf16)in[(size_t)rowBase * E_DIM + idx];
  }
  __syncthreads();
  if (tid < 64) {
    float m = 0.f;
    for (int e = 0; e < E_DIM; ++e) m += (float)sA[tid][e];
    m *= (1.0f / E_DIM);
    float var = 0.f;
    for (int e = 0; e < E_DIM; ++e) { float d = (float)sA[tid][e] - m; var += d * d; }
    var *= (1.0f / E_DIM);
    float rstd = rsqrtf(var + 1e-5f);
    for (int e = 0; e < E_DIM; ++e)
      sA[tid][e] = (__bf16)(((float)sA[tid][e] - m) * rstd * lnw[e]);
  }
  __syncthreads();

  const int wave = tid >> 5;
  const int lane = tid & 31;
  const int hl = (lane >> 4) & 1;
  const int N = lane & 15;
  for (int t = 0; t < 2; ++t) {           // two N-tiles cover C = 32
    const int n0 = t * 16;
    v8f accg = {};
    v8f accl = {};
#pragma unroll
    for (int kk = 0; kk < 4; ++kk) {      // K = 128 in four 32-steps
      v16bf a  = frag_a_ld(&sA[wave * 16][kk * 32], E_DIM);
      v16bf bg = frag_b_ld(&sWg[n0][kk * 32], E_DIM);
      v16bf bl = frag_b_ld(&sWl[n0][kk * 32], E_DIM);
      accg = wmma_bf16(a, bg, accg);
      accl = wmma_bf16(a, bl, accl);
    }
#pragma unroll
    for (int r = 0; r < 8; ++r) {
      const int M = r + 8 * hl;
      const int c = n0 + N;
      const int grow = rowBase + wave * 16 + M;
      float val = sigmoidf_(accg[r]) * (accl[r] + lb[c]);
      if (mask) val *= mask[grow];
      const int p = grow / W, q = grow - p * W;
      const int x = swapxy ? q : p;
      const int y = swapxy ? p : q;
      out[(size_t)c * chanStride + (size_t)x * rowStride + y] = (__bf16)val;
    }
  }
}

// ---------------------------------------------------------------------------
// Kernel 2: g = sigmoid(ln(z) @ eg_w.T + eg_b), bf16 out [row][E].
// ---------------------------------------------------------------------------
__global__ void k_ln_sig_proj(const float* __restrict__ in,    // z
                              const float* __restrict__ lnw,
                              const float* __restrict__ egw,   // (E,E) = [n][k]
                              const float* __restrict__ egb,   // (E,)
                              __bf16* __restrict__ g_out) {
  __shared__ __bf16 sA[64][E_DIM];
  __shared__ __bf16 sW[E_DIM][E_DIM];
  const int tid = threadIdx.x;
  const int rowBase = blockIdx.x * 64;

  for (int idx = tid; idx < E_DIM * E_DIM; idx += 128)
    sW[idx >> 7][idx & 127] = (__bf16)egw[idx];
  for (int idx = tid; idx < 64 * E_DIM; idx += 128)
    sA[idx >> 7][idx & 127] = (__bf16)in[(size_t)rowBase * E_DIM + idx];
  __syncthreads();
  if (tid < 64) {
    float m = 0.f;
    for (int e = 0; e < E_DIM; ++e) m += (float)sA[tid][e];
    m *= (1.0f / E_DIM);
    float var = 0.f;
    for (int e = 0; e < E_DIM; ++e) { float d = (float)sA[tid][e] - m; var += d * d; }
    var *= (1.0f / E_DIM);
    float rstd = rsqrtf(var + 1e-5f);
    for (int e = 0; e < E_DIM; ++e)
      sA[tid][e] = (__bf16)(((float)sA[tid][e] - m) * rstd * lnw[e]);
  }
  __syncthreads();

  const int wave = tid >> 5;
  const int lane = tid & 31;
  const int hl = (lane >> 4) & 1;
  const int N = lane & 15;
  for (int t = 0; t < 8; ++t) {           // N = 128 in eight 16-tiles
    v8f acc = {};
#pragma unroll
    for (int kk = 0; kk < 4; ++kk) {
      v16bf a = frag_a_ld(&sA[wave * 16][kk * 32], E_DIM);
      v16bf b = frag_b_ld(&sW[t * 16][kk * 32], E_DIM);
      acc = wmma_bf16(a, b, acc);
    }
#pragma unroll
    for (int r = 0; r < 8; ++r) {
      const int M = r + 8 * hl;
      const int n = t * 16 + N;
      const int grow = rowBase + wave * 16 + M;
      g_out[(size_t)grow * E_DIM + n] = (__bf16)sigmoidf_(acc[r] + egb[n]);
    }
  }
}

// ---------------------------------------------------------------------------
// Kernel 3: per-channel triangle GEMMs, accumulated:
//   S[i][j][c] = sum_k PP[c][i][k]*AB1T[c][j][k]  (K = 768, 24 steps)
//             + sum_k AB2[c][i][k]*CPT[c][j][k]   (K = 128,  4 steps)
// grid = (i/64, j/64, c); 256 threads = 8 waves = 4x2 sub-tiles; each wave
// owns one 16-row A strip x two 16-col B strips (2 WMMAs / step).
// Double-buffered GLOBAL_LOAD_ASYNC_TO_LDS_B128 staging: stage s+1 is issued,
// then s_wait_asynccnt 2 retires stage s (per-wave in-order completion), so
// global->LDS traffic overlaps the WMMAs.
// ---------------------------------------------------------------------------
__global__ void k_triangle(const __bf16* __restrict__ PP,    // [32][768][768]
                           const __bf16* __restrict__ AB1T,  // [32][128][768]
                           const __bf16* __restrict__ AB2,   // [32][768][128]
                           const __bf16* __restrict__ CPT,   // [32][128][128]
                           float* __restrict__ S) {          // [98304][32]
  __shared__ __bf16 tA[2][64][32];
  __shared__ __bf16 tB[2][64][32];
  const int tid = threadIdx.x;
  const int c = blockIdx.z;
  const int iBase = blockIdx.x * 64;
  const int jBase = blockIdx.y * 64;
  const int wave = tid >> 5;
  const int lane = tid & 31;
  const int wi = wave >> 1;   // 0..3  (16-row A strip)
  const int wj = wave & 1;    // 0..1  (32-col B strip = two 16-col tiles)

  const __bf16* PPc  = PP   + (size_t)c * (NP_DIM * NP_DIM);
  const __bf16* AB1c = AB1T + (size_t)c * (NC_DIM * NP_DIM);
  const __bf16* AB2c = AB2  + (size_t)c * (NP_DIM * NC_DIM);
  const __bf16* CPTc = CPT  + (size_t)c * (NC_DIM * NC_DIM);

  const int row = tid >> 2;   // 0..63: A row (i) and B row (j)
  const int kc  = tid & 3;    // 16B chunk within the 32-wide K slice

  // step s in [0,24): block1 slice; s in [24,28): block2 slice
  auto stage = [&](int s, int buf) {
    const __bf16* Ab; const __bf16* Bb; int lda, ldb, k0;
    if (s < 24) { Ab = PPc;  lda = NP_DIM; Bb = AB1c; ldb = NP_DIM; k0 = s * 32; }
    else        { Ab = AB2c; lda = NC_DIM; Bb = CPTc; ldb = NC_DIM; k0 = (s - 24) * 32; }
    async_cp16(Ab + (size_t)(iBase + row) * lda + k0 + kc * 8, &tA[buf][row][kc * 8]);
    async_cp16(Bb + (size_t)(jBase + row) * ldb + k0 + kc * 8, &tB[buf][row][kc * 8]);
  };

  v8f acc0 = {};
  v8f acc1 = {};
  const int NSTEP = 28;
  stage(0, 0);
  for (int s = 0; s < NSTEP; ++s) {
    const int buf = s & 1;
    if (s + 1 < NSTEP) {
      stage(s + 1, buf ^ 1);   // prefetch next slice into the other buffer
      wait_async_le<2>();      // the 2 in flight are s+1's -> slice s landed
    } else {
      wait_async_le<0>();
    }
    __syncthreads();           // everyone's slice-s LDS writes visible
    v16bf a  = frag_a_ld(&tA[buf][wi * 16][0], 32);
    v16bf b0 = frag_b_ld(&tB[buf][wj * 32][0], 32);
    v16bf b1 = frag_b_ld(&tB[buf][wj * 32 + 16][0], 32);
    acc0 = wmma_bf16(a, b0, acc0);
    acc1 = wmma_bf16(a, b1, acc1);
    __syncthreads();           // reads done before buf is overwritten (s+2)
  }

  const int hl = (lane >> 4) & 1;
  const int N = lane & 15;
#pragma unroll
  for (int r = 0; r < 8; ++r) {
    const int gi = iBase + wi * 16 + r + 8 * hl;
    const int gj = jBase + wj * 32 + N;
    S[((size_t)gi * NC_DIM + gj) * C_DIM + c] = acc0[r];
    S[((size_t)gi * NC_DIM + gj + 16) * C_DIM + c] = acc1[r];
  }
}

// ---------------------------------------------------------------------------
// Kernel 4: out = g * (LN_c(S) @ las_w.T + las_b) * mask.  K = 32 (one WMMA).
// ---------------------------------------------------------------------------
__global__ void k_final(const float* __restrict__ S,      // [98304][32]
                        const float* __restrict__ lncw,   // (32,)
                        const float* __restrict__ lasw,   // (E,C) = [n][k]
                        const float* __restrict__ lasb,   // (E,)
                        const __bf16* __restrict__ G,     // [98304][128]
                        const float* __restrict__ mask,   // [98304]
                        float* __restrict__ out) {        // [98304][128]
  __shared__ __bf16 sS[64][C_DIM];
  __shared__ __bf16 sW[E_DIM][C_DIM];
  const int tid = threadIdx.x;
  const int base = blockIdx.x * 64;

  for (int idx = tid; idx < E_DIM * C_DIM; idx += 128)
    sW[idx >> 5][idx & 31] = (__bf16)lasw[idx];
  if (tid < 64) {
    const int pt = base + tid;
    float v[C_DIM];
    float m = 0.f;
#pragma unroll
    for (int cc = 0; cc < C_DIM; ++cc) { v[cc] = S[(size_t)pt * C_DIM + cc]; m += v[cc]; }
    m *= (1.0f / C_DIM);
    float var = 0.f;
#pragma unroll
    for (int cc = 0; cc < C_DIM; ++cc) { const float d = v[cc] - m; var += d * d; }
    var *= (1.0f / C_DIM);
    const float rstd = rsqrtf(var + 1e-5f);
#pragma unroll
    for (int cc = 0; cc < C_DIM; ++cc)
      sS[tid][cc] = (__bf16)((v[cc] - m) * rstd * lncw[cc]);
  }
  __syncthreads();

  const int wave = tid >> 5;
  const int lane = tid & 31;
  const int hl = (lane >> 4) & 1;
  const int N = lane & 15;
  v16bf a = frag_a_ld(&sS[wave * 16][0], C_DIM);
  for (int t = 0; t < 8; ++t) {
    v8f acc = {};
    v16bf b = frag_b_ld(&sW[t * 16][0], C_DIM);
    acc = wmma_bf16(a, b, acc);
#pragma unroll
    for (int r = 0; r < 8; ++r) {
      const int pt = base + wave * 16 + r + 8 * hl;
      const int e = t * 16 + N;
      const float val = (acc[r] + lasb[e]) * (float)G[(size_t)pt * E_DIM + e] * mask[pt];
      __builtin_nontemporal_store(val, &out[(size_t)pt * E_DIM + e]);  // write-once
    }
  }
}

// ---------------------------------------------------------------------------
extern "C" void kernel_launch(void* const* d_in, const int* in_sizes, int n_in,
                              void* d_out, int out_size, void* d_ws, size_t ws_size,
                              hipStream_t stream) {
  (void)in_sizes; (void)n_in; (void)out_size; (void)ws_size;
  const float* z    = (const float*)d_in[0];
  const float* pp   = (const float*)d_in[1];
  const float* cp   = (const float*)d_in[2];
  const float* zm   = (const float*)d_in[3];
  const float* lnw  = (const float*)d_in[4];
  const float* lncw = (const float*)d_in[5];
  const float* gl1  = (const float*)d_in[6];
  const float* gl2  = (const float*)d_in[7];
  const float* l1w  = (const float*)d_in[8];
  const float* l1b  = (const float*)d_in[9];
  const float* l2w  = (const float*)d_in[10];
  const float* l2b  = (const float*)d_in[11];
  const float* egw  = (const float*)d_in[12];
  const float* egb  = (const float*)d_in[13];
  const float* lasw = (const float*)d_in[14];
  const float* lasb = (const float*)d_in[15];

  char* ws = (char*)d_ws;
  size_t o = 0;
  __bf16* PP   = (__bf16*)(ws + o); o += (size_t)C_DIM * NP_DIM * NP_DIM * 2;  // 37.7 MB
  __bf16* AB1T = (__bf16*)(ws + o); o += (size_t)C_DIM * NC_DIM * NP_DIM * 2;  //  6.3 MB
  __bf16* AB2  = (__bf16*)(ws + o); o += (size_t)C_DIM * NP_DIM * NC_DIM * 2;  //  6.3 MB
  __bf16* CPT  = (__bf16*)(ws + o); o += (size_t)C_DIM * NC_DIM * NC_DIM * 2;  //  1.0 MB
  __bf16* G    = (__bf16*)(ws + o); o += (size_t)NPTS * E_DIM * 2;             // 25.2 MB
  float*  S    = (float*)(ws + o);                                             // 12.6 MB

  // --- all three z readers first, so z (50 MB) stays L2-hot ---
  // ab1 = sig(z@gl1)*(z@l1+b1)*mask  -> AB1T[c][nc][np]   (B of block1, [N][K])
  k_ln_gate_proj<<<NPTS / 64, 128, 0, stream>>>(
      z, zm, lnw, gl1, l1w, l1b, AB1T, NC_DIM, NC_DIM * NP_DIM, NP_DIM, 1, 0);
  // ab2 -> AB2[c][np][nc]                                  (A of block2, [M][K])
  k_ln_gate_proj<<<NPTS / 64, 128, 0, stream>>>(
      z, zm, lnw, gl2, l2w, l2b, AB2, NC_DIM, NP_DIM * NC_DIM, NC_DIM, 0, 0);
  // g = sigmoid(ln(z) @ eg_w.T + eg_b)
  k_ln_sig_proj<<<NPTS / 64, 128, 0, stream>>>(z, lnw, egw, egb, G);
  // --- read-once 302 MB protein_pair stream, nontemporal ---
  // pp proj (gl2/l2 pair) -> PP[c][i][k]                   (A of block1, [M][K])
  k_ln_gate_proj<<<(NP_DIM * NP_DIM) / 64, 128, 0, stream>>>(
      pp, nullptr, lnw, gl2, l2w, l2b, PP, NP_DIM, NP_DIM * NP_DIM, NP_DIM, 0, 1);
  // cp proj (gl1/l1 pair) -> CPT[c][j][k]                  (B of block2, [N][K])
  k_ln_gate_proj<<<(NC_DIM * NC_DIM) / 64, 128, 0, stream>>>(
      cp, nullptr, lnw, gl1, l1w, l1b, CPT, NC_DIM, NC_DIM * NC_DIM, NC_DIM, 0, 0);
  // triangle updates (block1 + block2) -> S[point][c]
  k_triangle<<<dim3(NP_DIM / 64, NC_DIM / 64, C_DIM), 256, 0, stream>>>(
      PP, AB1T, AB2, CPT, S);
  // out = g * (LN_c(S) @ las_w.T + las_b) * mask
  k_final<<<NPTS / 64, 128, 0, stream>>>(S, lncw, lasw, lasb, G, zm, (float*)d_out);
}
